// ptTransformerBlock_CrossAttn_66099546685915
// MI455X (gfx1250) — matrix operand order repaired and checked by reading, not compiled, not run on hardware
//
#include <hip/hip_runtime.h>
#include <cstddef>

// Problem constants (from reference)
#define BB 4
#define NN 8192
#define MM 2048
#define DP 128
#define DM 256
#define KK 16
#define QDIM (3 + DP)   // 131 floats per query row

typedef __attribute__((ext_vector_type(16))) _Float16 v16h;
typedef __attribute__((ext_vector_type(8)))  _Float16 half8;
typedef __attribute__((ext_vector_type(8)))  float    v8f;

// Packed f16 weight offsets (in halves) within workspace weight region
#define OFF_FC1 0            // 128x256 -> 32768
#define OFF_WK  32768        // 256x256 -> 65536
#define OFF_WV  98304
#define OFF_D2  163840
#define OFF_G1  229376
#define OFF_G2  294912
#define WPACK_HALVES 360448  // total halves

// Per-wave LDS slice layout (bytes)
#define L_XIN   0        // 16x128 f16  = 4096
#define L_XBUF  4096     // 16x256 f16  = 8192
#define L_HBUF  12288    // 16x256 f16  = 8192 (hpos, reused as h)
#define L_PBUF  20480    // 16x256 f16  = 8192 (pos_enc)
#define L_ABUF  28672    // 16x256 f16  = 8192
#define L_VBUF  36864    // 16x256 f16  = 8192
#define L_QROW  45056    // 256 f32     = 1024
#define L_RES   46080    // 256 f32     = 1024
#define L_DXYZ  47104    // 16x4 f32    = 256
#define L_IDX   47360    // 16 int      = 64
#define PW_BYTES 47616   // per-wave slice (16B aligned)

__device__ __forceinline__ void wave_sync() {
    __builtin_amdgcn_wave_barrier();
}

// Load A-fragment (16x32 f16 tile) from row-major LDS buffer.
// Lane l<16: row l, K = [0..7, 16..23]; lane>=16: row l-16, K = [8..15, 24..31]
__device__ __forceinline__ v16h load_a_frag(const _Float16* buf, int stride, int kstep, int lane) {
    int half = lane >> 4;
    int row  = lane & 15;
    const _Float16* p = buf + row * stride + kstep * 32 + half * 8;
    half8 lo = *(const half8*)(p);
    half8 hi = *(const half8*)(p + 16);
    v16h a;
#pragma unroll
    for (int i = 0; i < 8; i++) { a[i] = lo[i]; a[i + 8] = hi[i]; }
    return a;
}

__device__ __forceinline__ v8f wmma_f16(v16h a, v16h b, v8f c) {
    return __builtin_amdgcn_wmma_f32_16x16x32_f16(false, a, false, b, (short)0, c, false, false);
}

// Generic 16xKd @ KdxDM WMMA stage: Dst (f16, 16x256 row-major LDS) = act(A @ W + bias)
__device__ __forceinline__ void gemm_stage(const _Float16* A, int Kd,
                                           const _Float16* W, const float* bias,
                                           _Float16* Dst, int lane, bool relu) {
    const int ks    = Kd >> 5;
    const int col16 = lane & 15;
    const int rbase = (lane >> 4) * 8;
    for (int nt = 0; nt < 16; nt++) {
        if (nt < 15) __builtin_prefetch(W + (((nt + 1) * ks) << 9), 0, 2);
        v8f acc = {};
        for (int kstep = 0; kstep < ks; kstep++) {
            v16h a  = load_a_frag(A, Kd, kstep, lane);
            v16h bf = *(const v16h*)(W + (((nt * ks + kstep) << 9) + (lane << 4)));
            acc = wmma_f16(a, bf, acc);
        }
        int col  = nt * 16 + col16;
        float bv = bias ? bias[col] : 0.0f;
#pragma unroll
        for (int i = 0; i < 8; i++) {
            float v = acc[i] + bv;
            if (relu) v = fmaxf(v, 0.0f);
            Dst[(rbase + i) * DM + col] = (_Float16)v;
        }
    }
    wave_sync();
}

// ---------------------------------------------------------------------------
// Kernel 1: pack f32 weight (Kd x Nd) into WMMA B-fragment f16 layout
// ---------------------------------------------------------------------------
__global__ void pack_weight_kernel(const float* __restrict__ src, _Float16* __restrict__ dst,
                                   int Kd, int Nd) {
    int tid   = blockIdx.x * 256 + threadIdx.x;
    int total = Kd * Nd;
    if (tid >= total) return;
    int ks    = Kd >> 5;
    int j     = tid & 15;
    int lane  = (tid >> 4) & 31;
    int kstep = (tid >> 9) % ks;
    int ntile = tid / (512 * ks);
    int row = kstep * 32 + (lane >> 4) * 8 + (j & 7) + (j >> 3) * 16;   // K index
    int col = ntile * 16 + (lane & 15);            // N index
    dst[tid] = (_Float16)src[row * Nd + col];
}

// ---------------------------------------------------------------------------
// Kernel 2: KNN (one wave32 per query). Writes K=16 nearest indices, ascending.
// ---------------------------------------------------------------------------
__global__ __launch_bounds__(128) void knn_kernel(const float* __restrict__ xyz,
                                                  const float* __restrict__ query,
                                                  int* __restrict__ knn_idx) {
    int lane = threadIdx.x & 31;
    int wave = threadIdx.x >> 5;
    int qi   = blockIdx.x * 4 + wave;
    int b    = qi / MM;
    const float* qp = query + (size_t)qi * QDIM;
    float qx = qp[0], qy = qp[1], qz = qp[2];
    const float* xp = xyz + (size_t)b * NN * 3;

    float dl[KK];
    int   il[KK];
#pragma unroll
    for (int i = 0; i < KK; i++) { dl[i] = 3.0e38f; il[i] = 0; }
    float worst = 3.0e38f;
    int   wslot = 0;

    for (int p = lane; p < NN; p += 32) {
        float dx = qx - xp[p * 3 + 0];
        float dy = qy - xp[p * 3 + 1];
        float dz = qz - xp[p * 3 + 2];
        float d  = dx * dx + dy * dy + dz * dz;
        if (d < worst) {
#pragma unroll
            for (int j = 0; j < KK; j++) if (j == wslot) { dl[j] = d; il[j] = p; }
            worst = dl[0]; wslot = 0;
#pragma unroll
            for (int j = 1; j < KK; j++) if (dl[j] > worst) { worst = dl[j]; wslot = j; }
        }
    }

    int* outp = knn_idx + (size_t)qi * KK;
    for (int sel = 0; sel < KK; sel++) {
        float bd = dl[0]; int bs = 0; int bidx = il[0];
#pragma unroll
        for (int j = 1; j < KK; j++) if (dl[j] < bd) { bd = dl[j]; bs = j; bidx = il[j]; }
        int blane = lane;
#pragma unroll
        for (int off = 16; off > 0; off >>= 1) {
            float od = __shfl_xor(bd, off, 32);
            int   oi = __shfl_xor(bidx, off, 32);
            int   ol = __shfl_xor(blane, off, 32);
            if (od < bd || (od == bd && ol < blane)) { bd = od; bidx = oi; blane = ol; }
        }
        if (lane == blane) {
#pragma unroll
            for (int j = 0; j < KK; j++) if (j == bs) dl[j] = 3.0e38f;
        }
        if (lane == 0) outp[sel] = bidx;
    }
}

// ---------------------------------------------------------------------------
// Kernel 3: main fused block. One wave per query (16 neighbor rows = 1 M-tile).
// ---------------------------------------------------------------------------
__global__ __launch_bounds__(128) void pt_block_kernel(
    const float* __restrict__ xyz, const float* __restrict__ features,
    const float* __restrict__ query, const _Float16* __restrict__ wpack,
    const float* __restrict__ fc1_b, const float* __restrict__ fc2_w,
    const float* __restrict__ fc2_b, const float* __restrict__ d1_w,
    const float* __restrict__ d1_b, const float* __restrict__ d2_b,
    const float* __restrict__ g1_b, const float* __restrict__ g2_b,
    const float* __restrict__ wq, const int* __restrict__ knn_idx,
    float* __restrict__ out_res, float* __restrict__ out_attn) {

    extern __shared__ char smem[];
    int lane = threadIdx.x & 31;
    int wave = threadIdx.x >> 5;
    char* base = smem + (size_t)wave * PW_BYTES;
    _Float16* xin  = (_Float16*)(base + L_XIN);
    _Float16* xbuf = (_Float16*)(base + L_XBUF);
    _Float16* hbuf = (_Float16*)(base + L_HBUF);
    _Float16* pbuf = (_Float16*)(base + L_PBUF);
    _Float16* abuf = (_Float16*)(base + L_ABUF);
    _Float16* vbuf = (_Float16*)(base + L_VBUF);
    float* qrow   = (float*)(base + L_QROW);
    float* resrow = (float*)(base + L_RES);
    float* dxyz   = (float*)(base + L_DXYZ);
    int*   idxs   = (int*)(base + L_IDX);

    const _Float16* fc1p = wpack + OFF_FC1;
    const _Float16* wkp  = wpack + OFF_WK;
    const _Float16* wvp  = wpack + OFF_WV;
    const _Float16* d2p  = wpack + OFF_D2;
    const _Float16* g1p  = wpack + OFF_G1;
    const _Float16* g2p  = wpack + OFF_G2;

    int qi = blockIdx.x * 4 + wave;
    int b  = qi / MM;
    const float* qp  = query + (size_t)qi * QDIM;
    const int*   kip = knn_idx + (size_t)qi * KK;

    // ---- S0a: neighbor indices + delta xyz ----
    if (lane < KK) {
        int id = kip[lane];
        idxs[lane] = id;
        const float* pp = xyz + ((size_t)b * NN + id) * 3;
        dxyz[lane * 4 + 0] = qp[0] - pp[0];
        dxyz[lane * 4 + 1] = qp[1] - pp[1];
        dxyz[lane * 4 + 2] = qp[2] - pp[2];
    }
    wave_sync();

    // ---- S0b: gather neighbor features -> xin (16x128 f16) ----
    for (int r = 0; r < KK; r++) {
        const float* fp = features + ((size_t)b * NN + idxs[r]) * DP;
        for (int c = lane; c < DP; c += 32) xin[r * DP + c] = (_Float16)fp[c];
    }

    // ---- S0c: q row = query_f @ wq (1x256, f32 matvec) ----
    for (int c = lane; c < DM; c += 32) {
        float s = 0.0f;
        for (int kk = 0; kk < DP; kk++) s += qp[3 + kk] * wq[kk * DM + c];
        qrow[c] = s;
    }

    // ---- S0d: hpos = relu(delta @ d1_w + d1_b) (16x256 f16, tiny K=3) ----
    for (int t = lane; t < KK * DM; t += 32) {
        int r = t >> 8, c = t & 255;
        float s = d1_b[c] + dxyz[r * 4 + 0] * d1_w[0 * DM + c]
                          + dxyz[r * 4 + 1] * d1_w[1 * DM + c]
                          + dxyz[r * 4 + 2] * d1_w[2 * DM + c];
        hbuf[t] = (_Float16)fmaxf(s, 0.0f);
    }
    wave_sync();

    // ---- S1: x = xin @ fc1 + fc1_b (WMMA, K=128) ----
    gemm_stage(xin, DP, fc1p, fc1_b, xbuf, lane, false);

    // ---- S2: pos = hpos @ d2 + d2_b (WMMA, K=256) ----
    gemm_stage(hbuf, DM, d2p, d2_b, pbuf, lane, false);

    // ---- S3: fused k_mat = x@wk, v = x@wv, a = q - k_mat + pos ----
    {
        const int col16 = lane & 15, rbase = (lane >> 4) * 8;
        for (int nt = 0; nt < 16; nt++) {
            v8f ak = {}, av = {};
            for (int kstep = 0; kstep < 8; kstep++) {
                v16h a  = load_a_frag(xbuf, DM, kstep, lane);
                v16h bk = *(const v16h*)(wkp + (((nt * 8 + kstep) << 9) + (lane << 4)));
                v16h bv = *(const v16h*)(wvp + (((nt * 8 + kstep) << 9) + (lane << 4)));
                ak = wmma_f16(a, bk, ak);
                av = wmma_f16(a, bv, av);
            }
            int col  = nt * 16 + col16;
            float qc = qrow[col];
#pragma unroll
            for (int i = 0; i < 8; i++) {
                int r = rbase + i;
                float pv = (float)pbuf[r * DM + col];
                abuf[r * DM + col] = (_Float16)(qc - ak[i] + pv);
                vbuf[r * DM + col] = (_Float16)av[i];
            }
        }
        wave_sync();
    }

    // ---- S4: h = relu(a @ g1 + g1_b) (WMMA) ----
    gemm_stage(abuf, DM, g1p, g1_b, hbuf, lane, true);

    // ---- S5: attn = softmax_n((h@g2 + g2_b)/16); res = sum_n attn*(v+pos) ----
    {
        const int col16 = lane & 15, rbase = (lane >> 4) * 8;
        float* attn_out = out_attn + (size_t)qi * KK * DM;
        for (int nt = 0; nt < 16; nt++) {
            v8f acc = {};
            for (int kstep = 0; kstep < 8; kstep++) {
                v16h a  = load_a_frag(hbuf, DM, kstep, lane);
                v16h bf = *(const v16h*)(g2p + (((nt * 8 + kstep) << 9) + (lane << 4)));
                acc = wmma_f16(a, bf, acc);
            }
            int col  = nt * 16 + col16;
            float bv = g2_b[col];
            float sc[8], e[8];
            float mx = -3.0e38f;
#pragma unroll
            for (int i = 0; i < 8; i++) { sc[i] = (acc[i] + bv) * 0.0625f; mx = fmaxf(mx, sc[i]); }
            mx = fmaxf(mx, __shfl_xor(mx, 16, 32));   // 16 rows live in lanes c and c+16
            float sum = 0.0f;
#pragma unroll
            for (int i = 0; i < 8; i++) { e[i] = __expf(sc[i] - mx); sum += e[i]; }
            sum += __shfl_xor(sum, 16, 32);
            float inv = 1.0f / sum;
            float part = 0.0f;
#pragma unroll
            for (int i = 0; i < 8; i++) {
                float at = e[i] * inv;
                int r = rbase + i;
                attn_out[r * DM + col] = at;
                part += at * ((float)vbuf[r * DM + col] + (float)pbuf[r * DM + col]);
            }
            part += __shfl_xor(part, 16, 32);
            if (lane < 16) resrow[col] = part;
        }
        wave_sync();
    }

    // ---- S6: out = resrow @ fc2 + fc2_b + query_f (1x128 matvec) ----
    {
        const float* qf = qp + 3;
        float* orow = out_res + (size_t)qi * DP;
        for (int c = lane; c < DP; c += 32) {
            float s = fc2_b[c] + qf[c];
            for (int kk = 0; kk < DM; kk++) s += resrow[kk] * fc2_w[kk * DP + c];
            orow[c] = s;
        }
    }
}

// ---------------------------------------------------------------------------
extern "C" void kernel_launch(void* const* d_in, const int* in_sizes, int n_in,
                              void* d_out, int out_size, void* d_ws, size_t ws_size,
                              hipStream_t stream) {
    const float* xyz      = (const float*)d_in[0];
    const float* features = (const float*)d_in[1];
    const float* query    = (const float*)d_in[2];
    const float* fc1_w    = (const float*)d_in[3];
    const float* fc1_b    = (const float*)d_in[4];
    const float* fc2_w    = (const float*)d_in[5];
    const float* fc2_b    = (const float*)d_in[6];
    const float* d1_w     = (const float*)d_in[7];
    const float* d1_b     = (const float*)d_in[8];
    const float* d2_w     = (const float*)d_in[9];
    const float* d2_b     = (const float*)d_in[10];
    const float* g1_w     = (const float*)d_in[11];
    const float* g1_b     = (const float*)d_in[12];
    const float* g2_w     = (const float*)d_in[13];
    const float* g2_b     = (const float*)d_in[14];
    const float* wq       = (const float*)d_in[15];
    const float* wk       = (const float*)d_in[16];
    const float* wv       = (const float*)d_in[17];
    (void)in_sizes; (void)n_in; (void)out_size; (void)ws_size;

    // Workspace: [0, 512KB) knn indices; then packed f16 weights (~720KB)
    int* knn = (int*)d_ws;
    _Float16* wpack = (_Float16*)((char*)d_ws + (size_t)BB * MM * KK * sizeof(int));

    // Pack weights into WMMA B-fragment layout (once per launch; deterministic)
    pack_weight_kernel<<<(32768 + 255) / 256, 256, 0, stream>>>(fc1_w, wpack + OFF_FC1, 128, 256);
    pack_weight_kernel<<<(65536 + 255) / 256, 256, 0, stream>>>(wk,   wpack + OFF_WK, 256, 256);
    pack_weight_kernel<<<(65536 + 255) / 256, 256, 0, stream>>>(wv,   wpack + OFF_WV, 256, 256);
    pack_weight_kernel<<<(65536 + 255) / 256, 256, 0, stream>>>(d2_w, wpack + OFF_D2, 256, 256);
    pack_weight_kernel<<<(65536 + 255) / 256, 256, 0, stream>>>(g1_w, wpack + OFF_G1, 256, 256);
    pack_weight_kernel<<<(65536 + 255) / 256, 256, 0, stream>>>(g2_w, wpack + OFF_G2, 256, 256);

    // KNN: one wave per query, 4 waves per block
    knn_kernel<<<BB * MM / 4, 128, 0, stream>>>(xyz, query, knn);

    // Main fused block: one wave per query, 4 waves per block, ~186KB dyn LDS
    float* out_res  = (float*)d_out;
    float* out_attn = out_res + (size_t)BB * MM * DP;
    size_t lds_bytes = (size_t)4 * PW_BYTES;
    pt_block_kernel<<<BB * MM / 4, 128, lds_bytes, stream>>>(
        xyz, features, query, wpack, fc1_b, fc2_w, fc2_b, d1_w, d1_b,
        d2_b, g1_b, g2_b, wq, knn, out_res, out_attn);
}
